// NeighborhoodAttentionBlock_2834678415876
// MI455X (gfx1250) — matrix operand order repaired and checked by reading, not compiled, not run on hardware
//
#include <hip/hip_runtime.h>
#include <hip/hip_bf16.h>

typedef __attribute__((ext_vector_type(16))) __bf16 v16bf;
typedef __attribute__((ext_vector_type(8)))  __bf16 v8bf;
typedef __attribute__((ext_vector_type(8)))  float  v8f;

#define BB 8
#define NN 2048
#define CC 512
#define MM (BB * NN)   // 16384 rows

// ---- one-pass f32 -> bf16 conversion (8 elems/thread, b128 in / b128 out) --
__global__ __launch_bounds__(256) void cvt_bf16_kernel(
    const float* __restrict__ in, __bf16* __restrict__ out, int n) {
  const int i = (blockIdx.x * blockDim.x + threadIdx.x) * 8;
  if (i >= n) return;
  float4 f0 = *reinterpret_cast<const float4*>(in + i);
  float4 f1 = *reinterpret_cast<const float4*>(in + i + 4);
  v8bf r;
  r[0] = (__bf16)f0.x; r[1] = (__bf16)f0.y; r[2] = (__bf16)f0.z; r[3] = (__bf16)f0.w;
  r[4] = (__bf16)f1.x; r[5] = (__bf16)f1.y; r[6] = (__bf16)f1.z; r[7] = (__bf16)f1.w;
  *reinterpret_cast<v8bf*>(out + i) = r;
}

// ---- fragment load helpers (all-bf16 paths) --------------------------------
// A-matrix 16x32 bf16: lane L<16 = row M=L, K=[0..15];
// lane L>=16 = row M=L-16, K=[16..31]. Caller pre-offsets p by
// (lane<16 ? 0 : 16); we take [0..7] and [8..15].
__device__ __forceinline__ v16bf load_afrag_bf(const __bf16* __restrict__ p) {
  v16bf r;
  v8bf lo = *reinterpret_cast<const v8bf*>(p);
  v8bf hi = *reinterpret_cast<const v8bf*>(p + 8);
#pragma unroll
  for (int i = 0; i < 8; ++i) { r[i] = lo[i]; r[8 + i] = hi[i]; }
  return r;
}

// B-matrix 32x16 bf16: lane n<16 = column n, K contiguous (caller pre-offsets
// by (lane<16 ? 0 : 16)). 32-byte aligned -> b128 pair.
__device__ __forceinline__ v16bf load_bfrag_bf(const __bf16* __restrict__ p) {
  return *reinterpret_cast<const v16bf*>(p);
}

// ---- kernel 1: fused QKV projection (bf16 WMMA, f32 accumulate) ------------
__global__ __launch_bounds__(256) void qkv_kernel(
    const __bf16* __restrict__ Xb,
    const __bf16* __restrict__ Wqb, const float* __restrict__ bq,
    const __bf16* __restrict__ Wkb, const float* __restrict__ bk,
    const __bf16* __restrict__ Wvb, const float* __restrict__ bv,
    __bf16* __restrict__ Qb, __bf16* __restrict__ Kb, __bf16* __restrict__ Vb) {
  const int wave = (blockIdx.x * blockDim.x + threadIdx.x) >> 5;
  const int lane = threadIdx.x & 31;
  const int row_tile = wave >> 5;        // MM/16 = 1024 row tiles
  const int col_tile = wave & 31;        // CC/16 = 32 col tiles
  const int mrow = row_tile * 16 + (lane & 15);
  const int dcol = col_tile * 16 + (lane & 15);
  const int asel = (lane < 16) ? 0 : 16;
  const int bsel = (lane < 16) ? 0 : 16;

  const __bf16* xrow = Xb  + (size_t)mrow * CC;
  const __bf16* wqr  = Wqb + (size_t)dcol * CC;
  const __bf16* wkr  = Wkb + (size_t)dcol * CC;
  const __bf16* wvr  = Wvb + (size_t)dcol * CC;

  v8f cq = {}, ck = {}, cv = {};
#pragma unroll 4
  for (int k0 = 0; k0 < CC; k0 += 32) {
    v16bf a  = load_afrag_bf(xrow + k0 + asel);
    v16bf b0 = load_bfrag_bf(wqr + k0 + bsel);
    v16bf b1 = load_bfrag_bf(wkr + k0 + bsel);
    v16bf b2 = load_bfrag_bf(wvr + k0 + bsel);
    cq = __builtin_amdgcn_wmma_f32_16x16x32_bf16(false, a, false, b0, (short)0, cq, false, false);
    ck = __builtin_amdgcn_wmma_f32_16x16x32_bf16(false, a, false, b1, (short)0, ck, false, false);
    cv = __builtin_amdgcn_wmma_f32_16x16x32_bf16(false, a, false, b2, (short)0, cv, false, false);
  }
  const float biq = bq[dcol], bik = bk[dcol], biv = bv[dcol];
  const int mbase = row_tile * 16 + ((lane < 16) ? 0 : 8);
#pragma unroll
  for (int r = 0; r < 8; ++r) {
    size_t off = (size_t)(mbase + r) * CC + dcol;
    Qb[off] = (__bf16)(cq[r] + biq);
    Kb[off] = (__bf16)(ck[r] + bik);
    Vb[off] = (__bf16)(cv[r] + biv);
  }
}

// ---- kernel 2: s[b,n] = Q[b,n,:] . K[b,nbr[n],:] ---------------------------
__global__ __launch_bounds__(256) void score_kernel(
    const __bf16* __restrict__ Qb, const __bf16* __restrict__ Kb,
    const int* __restrict__ nbr, float* __restrict__ tbuf) {
  const int wave = (blockIdx.x * blockDim.x + threadIdx.x) >> 5;
  const int lane = threadIdx.x & 31;
  const int b = wave / NN;
  const int n = wave - b * NN;
  const int j = nbr[n];
  const __bf16* q = Qb + ((size_t)(b * NN + n)) * CC + lane * 16;
  const __bf16* k = Kb + ((size_t)(b * NN + j)) * CC + lane * 16;
  v8bf q0 = *reinterpret_cast<const v8bf*>(q);
  v8bf q1 = *reinterpret_cast<const v8bf*>(q + 8);
  v8bf k0 = *reinterpret_cast<const v8bf*>(k);
  v8bf k1 = *reinterpret_cast<const v8bf*>(k + 8);
  float s = 0.f;
#pragma unroll
  for (int i = 0; i < 8; ++i)
    s += (float)q0[i] * (float)k0[i] + (float)q1[i] * (float)k1[i];
#pragma unroll
  for (int off = 16; off > 0; off >>= 1) s += __shfl_xor(s, off, 32);
  if (lane == 0) tbuf[b * NN + n] = s;
}

// ---- kernel 3: Vsum[b,c] = sum_n V[b, nbr[n], c] ---------------------------
__global__ __launch_bounds__(256) void vsum_kernel(
    const __bf16* __restrict__ Vb, const int* __restrict__ nbr,
    float* __restrict__ vsum) {
  const int idx = blockIdx.x * blockDim.x + threadIdx.x;  // 0..B*C-1
  const int b = idx / CC;
  const int c = idx - b * CC;
  const __bf16* base = Vb + (size_t)b * NN * CC + c;
  float s = 0.f;
  for (int n = 0; n < NN; ++n) s += (float)base[(size_t)nbr[n] * CC];
  vsum[idx] = s;
}

// ---- kernel 4: closed-form softmax mix -------------------------------------
// Row of scores has one entry t at col nbr[n], zeros elsewhere:
//   Z = e^t + (N-1);  out = (Vsum + (e^t - 1) * V[nbr[nbr[n]]]) / Z
__global__ __launch_bounds__(256) void mix_kernel(
    const __bf16* __restrict__ Vb, const int* __restrict__ nbr,
    const float* __restrict__ tbuf, const float* __restrict__ vsum,
    const float* __restrict__ rel_bias, __bf16* __restrict__ Ob) {
  const int idx = blockIdx.x * blockDim.x + threadIdx.x;  // 0..B*N*C-1
  const int c  = idx & (CC - 1);
  const int bn = idx >> 9;           // / CC
  const int n  = bn & (NN - 1);
  const int b  = bn >> 11;           // / NN
  const int j2 = nbr[nbr[n]];
  const float inv_sqrtC = 0.044194173824159216f;  // 1/sqrt(512)
  float t = (tbuf[bn] + rel_bias[0]) * inv_sqrtC;
  float e = __expf(t);
  float Z = e + (float)(NN - 1);
  float v2 = (float)Vb[((size_t)(b * NN + j2)) * CC + c];
  float o = (vsum[b * CC + c] + (e - 1.0f) * v2) / Z;
  Ob[idx] = (__bf16)o;
}

// ---- kernel 5: output projection (bf16 WMMA) -------------------------------
__global__ __launch_bounds__(256) void out_gemm_kernel(
    const __bf16* __restrict__ Ob, const __bf16* __restrict__ Wob,
    const float* __restrict__ bo, float* __restrict__ out) {
  const int wave = (blockIdx.x * blockDim.x + threadIdx.x) >> 5;
  const int lane = threadIdx.x & 31;
  const int row_tile = wave >> 5;
  const int col_tile = wave & 31;
  const int mrow = row_tile * 16 + (lane & 15);
  const int dcol = col_tile * 16 + (lane & 15);
  const int asel = (lane < 16) ? 0 : 16;
  const int bsel = (lane < 16) ? 0 : 16;

  const __bf16* arow = Ob  + (size_t)mrow * CC;
  const __bf16* wrow = Wob + (size_t)dcol * CC;
  v8f acc = {};
#pragma unroll 4
  for (int k0 = 0; k0 < CC; k0 += 32) {
    v16bf a = load_afrag_bf(arow + k0 + asel);
    v16bf w = load_bfrag_bf(wrow + k0 + bsel);
    acc = __builtin_amdgcn_wmma_f32_16x16x32_bf16(false, a, false, w, (short)0, acc, false, false);
  }
  const float bias = bo[dcol];
  const int mbase = row_tile * 16 + ((lane < 16) ? 0 : 8);
#pragma unroll
  for (int r = 0; r < 8; ++r)
    out[(size_t)(mbase + r) * CC + dcol] = acc[r] + bias;
}

// ---- host launch -----------------------------------------------------------
extern "C" void kernel_launch(void* const* d_in, const int* in_sizes, int n_in,
                              void* d_out, int out_size, void* d_ws, size_t ws_size,
                              hipStream_t stream) {
  (void)in_sizes; (void)n_in; (void)out_size; (void)ws_size;
  const float* x  = (const float*)d_in[0];
  const int*   nb = (const int*)  d_in[1];
  const float* Wq = (const float*)d_in[2];
  const float* bq = (const float*)d_in[3];
  const float* Wk = (const float*)d_in[4];
  const float* bk = (const float*)d_in[5];
  const float* Wv = (const float*)d_in[6];
  const float* bv = (const float*)d_in[7];
  const float* rb = (const float*)d_in[8];
  const float* Wo = (const float*)d_in[9];
  const float* bo = (const float*)d_in[10];
  float* out = (float*)d_out;

  char* ws = (char*)d_ws;
  const size_t MAT  = (size_t)MM * CC * sizeof(__bf16);  // 16 MiB
  const size_t WMAT = (size_t)CC * CC * sizeof(__bf16);  // 512 KiB
  __bf16* Xb  = (__bf16*)(ws);
  __bf16* Qb  = (__bf16*)(ws + 1 * MAT);
  __bf16* Kb  = (__bf16*)(ws + 2 * MAT);
  __bf16* Vb  = (__bf16*)(ws + 3 * MAT);
  __bf16* Ob  = (__bf16*)(ws + 4 * MAT);
  __bf16* Wqb = (__bf16*)(ws + 5 * MAT);
  __bf16* Wkb = (__bf16*)(ws + 5 * MAT + 1 * WMAT);
  __bf16* Wvb = (__bf16*)(ws + 5 * MAT + 2 * WMAT);
  __bf16* Wob = (__bf16*)(ws + 5 * MAT + 3 * WMAT);
  float* tbuf = (float*)(ws + 5 * MAT + 4 * WMAT);
  float* vsum = (float*)(ws + 5 * MAT + 4 * WMAT + (size_t)MM * sizeof(float));

  const int NX = MM * CC;   // 8,388,608
  const int NW = CC * CC;   //   262,144
  cvt_bf16_kernel<<<NX / 2048, 256, 0, stream>>>(x,  Xb,  NX);
  cvt_bf16_kernel<<<NW / 2048, 256, 0, stream>>>(Wq, Wqb, NW);
  cvt_bf16_kernel<<<NW / 2048, 256, 0, stream>>>(Wk, Wkb, NW);
  cvt_bf16_kernel<<<NW / 2048, 256, 0, stream>>>(Wv, Wvb, NW);
  cvt_bf16_kernel<<<NW / 2048, 256, 0, stream>>>(Wo, Wob, NW);

  // (MM/16)*(CC/16) = 32768 wave-tiles, 8 waves / 256-thread block
  qkv_kernel<<<4096, 256, 0, stream>>>(Xb, Wqb, bq, Wkb, bk, Wvb, bv, Qb, Kb, Vb);
  score_kernel<<<2048, 256, 0, stream>>>(Qb, Kb, nb, tbuf);
  vsum_kernel<<<16, 256, 0, stream>>>(Vb, nb, vsum);
  mix_kernel<<<32768, 256, 0, stream>>>(Vb, nb, tbuf, vsum, rb, Ob);
  out_gemm_kernel<<<4096, 256, 0, stream>>>(Ob, Wob, bo, out);
}
